// Bayesian_NN_Moe_75522704933183
// MI455X (gfx1250) — matrix operand order repaired and checked by reading, not compiled, not run on hardware
//
#include <hip/hip_runtime.h>

// ---------------- problem constants ----------------
constexpr int B = 2048, E = 16, F = 512, H = 1024, C = 512;
constexpr int TOPK = 4;
constexpr int TM = 64;            // tokens per gemm block
constexpr int A_STRIDE  = F + 16; // padded bf16 row stride for h tile
constexpr int AC_STRIDE = H + 16; // padded bf16 row stride for activation tile

// weight stage: 128 cols x 128 K bf16, TDM pad: 64 dwords data + 4 dwords pad
constexpr int STG_ROW_ELEMS = 136;            // 272 bytes per row
constexpr int STG_BYTES     = 128 * 272;      // 34816 per buffer

constexpr int OFF_A   = 0;
constexpr int OFF_AC  = TM * A_STRIDE * 2;
constexpr int OFF_TOK = OFF_AC + TM * AC_STRIDE * 2;
constexpr int OFF_STG = OFF_TOK + 512;
constexpr int SMEM_TOTAL = OFF_STG + 2 * STG_BYTES;

typedef __bf16 bf16_t;
typedef __attribute__((ext_vector_type(16))) __bf16 v16bf;
typedef __attribute__((ext_vector_type(8)))  float  v8f;
typedef __attribute__((ext_vector_type(4)))  unsigned int v4u;
typedef __attribute__((ext_vector_type(4)))  unsigned int u32x4;
typedef __attribute__((ext_vector_type(8)))  int i32x8;
typedef __attribute__((ext_vector_type(4)))  int i32x4;

union Frag { v4u u[2]; v16bf bf; };

__device__ __forceinline__ void atomicAddF(float* p, float v) {
    __hip_atomic_fetch_add(p, v, __ATOMIC_RELAXED, __HIP_MEMORY_SCOPE_AGENT);
}

// ---- TDM: DMA one 128(rows) x 128(K) bf16 tile global -> LDS, row pitch 272B ----
__device__ __forceinline__ void tdm_load_tile(unsigned lds_addr, const void* gptr,
                                              unsigned stride_elems) {
    unsigned long long ga = (unsigned long long)gptr;
    u32x4 g0 = { 1u,                                   // count=1 (valid descriptor)
                 lds_addr,                             // LDS byte address
                 (unsigned)ga,                         // global_addr[31:0]
                 (unsigned)((ga >> 32) & 0x01FFFFFFu)  // global_addr[56:32]
                   | 0x80000000u };                    // type=2 ("image")
    i32x8 g1 = { (int)((1u << 16)                      // data_size = 2 bytes
                     | (1u << 20)                      // pad_enable
                     | (5u << 22)                      // pad_interval: 64 dwords
                     | (3u << 25)),                    // pad_amount: 4 dwords
                 0,                                    // abar addr / tensor_dim0 lo
                 0x4000,                               // tensor_dim0 hi (big)
                 (int)(0x4000u | (128u << 16)),        // tensor_dim1 hi | tile_dim0=128
                 128,                                  // tile_dim1=128, tile_dim2=0
                 (int)stride_elems,                    // tensor_dim0_stride[31:0]
                 0, 0 };
    i32x4 z4 = { 0, 0, 0, 0 };
#if defined(__clang_major__) && (__clang_major__ >= 23)
    i32x8 z8 = { 0, 0, 0, 0, 0, 0, 0, 0 };
    __builtin_amdgcn_tensor_load_to_lds(g0, g1, z4, z4, z8, 0);
#else
    __builtin_amdgcn_tensor_load_to_lds(g0, g1, z4, z4, 0);
#endif
}

// ---------------- prep: fp32 -> bf16 copy of h ----------------
__global__ void cvt_h_kernel(const float* __restrict__ h, bf16_t* __restrict__ o, int n) {
    int i = blockIdx.x * 256 + threadIdx.x;
    if (i < n) o[i] = (bf16_t)h[i];
}

// ---------------- prep: per-expert transpose + convert  [R][Cc]f32 -> [Cc][R]bf16 ----
__global__ void transpose_cvt_kernel(const float* __restrict__ in, bf16_t* __restrict__ out,
                                     int R, int Cc) {
    __shared__ float tile[32][33];
    const int e = blockIdx.z;
    const float* src = in + (size_t)e * R * Cc;
    bf16_t* dst = out + (size_t)e * R * Cc;
    const int c0 = blockIdx.x * 32, r0 = blockIdx.y * 32;
    const int tx = threadIdx.x, ty = threadIdx.y;       // 32 x 8
#pragma unroll
    for (int i = 0; i < 32; i += 8)
        tile[ty + i][tx] = src[(size_t)(r0 + ty + i) * Cc + c0 + tx];
    __syncthreads();
#pragma unroll
    for (int i = 0; i < 32; i += 8)
        dst[(size_t)(c0 + ty + i) * R + r0 + tx] = (bf16_t)tile[tx][ty + i];
}

// ---------------- router: mu/var -> tilde -> softmax -> top-4 -> routing lists ----
__global__ void router_kernel(const float* __restrict__ h,
                              const float* __restrict__ W_mu,  const float* __restrict__ b_mu,
                              const float* __restrict__ W_lv,  const float* __restrict__ b_lv,
                              float* __restrict__ w_dense, int* __restrict__ counts,
                              int* __restrict__ lists) {
    const int tid = threadIdx.x;
    const int wv = tid >> 5, lane = tid & 31;
    const int b = blockIdx.x * 4 + wv;
    const int e = lane & 15, hf = lane >> 4;

    const float* hrow = h + (size_t)b * F;
    const float* wm = W_mu + (size_t)e * F;
    const float* wl = W_lv + (size_t)e * F;
    float mu = 0.f, var = 0.f;
    for (int f = hf * (F / 2); f < (hf + 1) * (F / 2); ++f) {
        float x = hrow[f];
        mu  = fmaf(x, wm[f], mu);
        var = fmaf(x * x, __expf(wl[f]), var);
    }
    mu  += __shfl_xor(mu, 16);
    var += __shfl_xor(var, 16);
    mu  += b_mu[e];
    var += __expf(b_lv[e]);
    var = fmaxf(var, 1e-12f);
    float tilde = mu * rsqrtf(1.0f + 0.39269908169872414f * var);  // pi/8

    float tl[16];
#pragma unroll
    for (int i = 0; i < 16; ++i) tl[i] = __shfl(tilde, i);

    if (lane == 0) {
        float mx = tl[0];
#pragma unroll
        for (int i = 1; i < 16; ++i) mx = fmaxf(mx, tl[i]);
        float pr[16], s = 0.f;
#pragma unroll
        for (int i = 0; i < 16; ++i) { pr[i] = __expf(tl[i] - mx); s += pr[i]; }
        float inv_s = 1.0f / s;

        bool used[16];
#pragma unroll
        for (int i = 0; i < 16; ++i) used[i] = false;
        int   isel[TOPK];
        float wsel[TOPK];
        float wsum = 0.f;
        for (int k = 0; k < TOPK; ++k) {
            float best = -3.4e38f; int bi = 0;
            for (int i = 0; i < 16; ++i)
                if (!used[i] && tl[i] > best) { best = tl[i]; bi = i; }
            used[bi] = true;
            isel[k] = bi;
            wsel[k] = pr[bi] * inv_s;
            wsum += wsel[k];
        }
        wsum = fmaxf(wsum, 1e-12f);
        for (int k = 0; k < TOPK; ++k) {
            float wval = wsel[k] / wsum;
            w_dense[b * E + isel[k]] = wval;
            int pos = atomicAdd(&counts[isel[k]], 1);
            lists[isel[k] * B + pos] = b;
        }
    }
}

// ---------------- fused expert MLP with TDM-staged weight tiles ----------------
__global__ __launch_bounds__(256)
void moe_gemm_kernel(const bf16_t* __restrict__ h_bf,
                     const bf16_t* __restrict__ W1t,   // [E][H][F] bf16
                     const float*  __restrict__ b1,
                     const bf16_t* __restrict__ W2t,   // [E][C][H] bf16
                     const float*  __restrict__ b2,
                     const float*  __restrict__ w_dense,
                     const int*    __restrict__ counts,
                     const int*    __restrict__ lists,
                     float* __restrict__ out) {
    extern __shared__ char smem[];
    bf16_t* A  = (bf16_t*)(smem + OFF_A);     // [TM][A_STRIDE]
    bf16_t* Ac = (bf16_t*)(smem + OFF_AC);    // [TM][AC_STRIDE]
    int*    tokS = (int*)(smem + OFF_TOK);
    float*  wS   = (float*)(smem + OFF_TOK + 256);
    bf16_t* stg0 = (bf16_t*)(smem + OFF_STG);
    bf16_t* stg1 = (bf16_t*)(smem + OFF_STG + STG_BYTES);
    // dynamic LDS begins right after static LDS (none here)
    const unsigned ldsBase = __builtin_amdgcn_groupstaticsize();
    const unsigned stgA0 = ldsBase + OFF_STG;
    const unsigned stgA1 = ldsBase + OFF_STG + STG_BYTES;

    const int e = blockIdx.y;
    const int cnt = counts[e];
    const int base = blockIdx.x * TM;
    if (base >= cnt) return;
    const int rows = (cnt - base < TM) ? (cnt - base) : TM;

    const int tid = threadIdx.x;
    const int wv = tid >> 5, lane = tid & 31;

    if (tid < TM) {
        if (tid < rows) {
            int tk = lists[e * B + base + tid];
            tokS[tid] = tk;
            wS[tid] = w_dense[tk * E + e];
        } else {
            tokS[tid] = 0;
            wS[tid] = 0.f;
        }
    }
    __syncthreads();

    const bf16_t* W1e = W1t + (size_t)e * H * F;
    const bf16_t* W2e = W2t + (size_t)e * C * H;

    // kick off first weight tile DMA (overlaps with the h gather below)
    if (wv == 0) tdm_load_tile(stgA0, W1e, F);

    // gather h rows (bf16) into LDS, 16B chunks
    for (int i = tid; i < TM * (F / 8); i += 256) {
        int r = i / (F / 8), ck = i % (F / 8);
        *(v4u*)(A + r * A_STRIDE + ck * 8) =
            *(const v4u*)(h_bf + (size_t)tokS[r] * F + ck * 8);
    }
    __syncthreads();

    const int m_base = (wv & 3) * 16;         // 4 M-slices of 16 rows
    const int n_half = (wv >> 2) * 64;        // 2 N-halves of 64 cols
    const int l15 = lane & 15;
    const int hiA = (lane >> 4) * 8;          // A fragment K sub-offset
    const int hiB = (lane >> 4) * 8;          // B fragment K sub-offset
    const int mrow = m_base + ((lane >> 4) << 3);

    // ---- GEMM1: a = relu(h @ W1 + b1); 8 n-chunks x 4 staged K tiles ----
    for (int nch = 0; nch < 8; ++nch) {
        const int n0 = nch << 7;
        v8f acc[4] = {};
        for (int ks = 0; ks < 4; ++ks) {
            const int s = nch * 4 + ks;
            if (wv == 0) {
                const int s1 = s + 1;
                if (s1 < 32) {
                    tdm_load_tile((s1 & 1) ? stgA1 : stgA0,
                                  W1e + (size_t)((s1 >> 2) << 7) * F + ((s1 & 3) << 7), F);
                    __builtin_amdgcn_s_wait_tensorcnt(1);   // tile s landed (in-order)
                } else {
                    __builtin_amdgcn_s_wait_tensorcnt(0);
                }
            }
            __syncthreads();                                // publish tile s

            const bf16_t* Wst = (s & 1) ? stg1 : stg0;
            const int k0 = ks << 7;
#pragma unroll
            for (int kk = 0; kk < 128; kk += 32) {
                Frag a;
                const bf16_t* ap = A + (m_base + l15) * A_STRIDE + k0 + kk + hiA;
                a.u[0] = *(const v4u*)ap;
                a.u[1] = *(const v4u*)(ap + 16);
                Frag bq[4];
#pragma unroll
                for (int j = 0; j < 4; ++j) {
                    const bf16_t* bp = Wst + (n_half + j * 16 + l15) * STG_ROW_ELEMS + kk + hiB;
                    bq[j].u[0] = *(const v4u*)bp;
                    bq[j].u[1] = *(const v4u*)(bp + 16);
                }
#pragma unroll
                for (int j = 0; j < 4; ++j)
                    acc[j] = __builtin_amdgcn_wmma_f32_16x16x32_bf16(
                        false, a.bf, false, bq[j].bf, (short)0, acc[j], false, false);
            }
            __syncthreads();                                // done reading tile s
        }
#pragma unroll
        for (int j = 0; j < 4; ++j) {
            const int col = n0 + n_half + j * 16 + l15;
            const float bias = b1[e * H + col];
#pragma unroll
            for (int r = 0; r < 8; ++r) {
                float v = acc[j][r] + bias;
                Ac[(mrow + r) * AC_STRIDE + col] = (bf16_t)fmaxf(v, 0.f);
            }
        }
    }

    // ---- GEMM2: y = a @ W2 + b2; out += w * y; 4 n-chunks x 8 staged K tiles ----
    if (wv == 0) tdm_load_tile(stgA0, W2e, H);
    for (int nch = 0; nch < 4; ++nch) {
        const int n0 = nch << 7;
        v8f acc[4] = {};
        for (int ks = 0; ks < 8; ++ks) {
            const int s = nch * 8 + ks;
            if (wv == 0) {
                const int s1 = s + 1;
                if (s1 < 32) {
                    tdm_load_tile((s1 & 1) ? stgA1 : stgA0,
                                  W2e + (size_t)((s1 >> 3) << 7) * H + ((s1 & 7) << 7), H);
                    __builtin_amdgcn_s_wait_tensorcnt(1);
                } else {
                    __builtin_amdgcn_s_wait_tensorcnt(0);
                }
            }
            __syncthreads();

            const bf16_t* Wst = (s & 1) ? stg1 : stg0;
            const int k0 = ks << 7;
#pragma unroll
            for (int kk = 0; kk < 128; kk += 32) {
                Frag a;
                const bf16_t* ap = Ac + (m_base + l15) * AC_STRIDE + k0 + kk + hiA;
                a.u[0] = *(const v4u*)ap;
                a.u[1] = *(const v4u*)(ap + 16);
                Frag bq[4];
#pragma unroll
                for (int j = 0; j < 4; ++j) {
                    const bf16_t* bp = Wst + (n_half + j * 16 + l15) * STG_ROW_ELEMS + kk + hiB;
                    bq[j].u[0] = *(const v4u*)bp;
                    bq[j].u[1] = *(const v4u*)(bp + 16);
                }
#pragma unroll
                for (int j = 0; j < 4; ++j)
                    acc[j] = __builtin_amdgcn_wmma_f32_16x16x32_bf16(
                        false, a.bf, false, bq[j].bf, (short)0, acc[j], false, false);
            }
            __syncthreads();
        }
#pragma unroll
        for (int j = 0; j < 4; ++j) {
            const int col = n0 + n_half + j * 16 + l15;
            const float bias = b2[e * C + col];
#pragma unroll
            for (int r = 0; r < 8; ++r) {
                const int m = mrow + r;
                if (m < rows) {
                    float v = (acc[j][r] + bias) * wS[m];
                    atomicAddF(&out[(size_t)tokS[m] * C + col], v);
                }
            }
        }
    }
}

// ---------------- workspace layout ----------------
constexpr size_t OFF_HBF  = 0;
constexpr size_t SZ_HBF   = (size_t)B * F * 2;
constexpr size_t OFF_W1T  = OFF_HBF + SZ_HBF;
constexpr size_t SZ_W1T   = (size_t)E * F * H * 2;
constexpr size_t OFF_W2T  = OFF_W1T + SZ_W1T;
constexpr size_t SZ_W2T   = (size_t)E * H * C * 2;
constexpr size_t OFF_WD   = OFF_W2T + SZ_W2T;
constexpr size_t SZ_WD    = (size_t)B * E * 4;
constexpr size_t OFF_CNT  = OFF_WD + SZ_WD;
constexpr size_t SZ_CNT   = 256;
constexpr size_t OFF_LST  = OFF_CNT + SZ_CNT;

extern "C" void kernel_launch(void* const* d_in, const int* in_sizes, int n_in,
                              void* d_out, int out_size, void* d_ws, size_t ws_size,
                              hipStream_t stream) {
    const float* h     = (const float*)d_in[0];
    const float* W_mu  = (const float*)d_in[1];
    const float* b_mu  = (const float*)d_in[2];
    const float* W_lv  = (const float*)d_in[3];
    const float* b_lv  = (const float*)d_in[4];
    const float* W1    = (const float*)d_in[5];
    const float* b1    = (const float*)d_in[6];
    const float* W2    = (const float*)d_in[7];
    const float* b2    = (const float*)d_in[8];
    float* out = (float*)d_out;

    char* ws = (char*)d_ws;
    bf16_t* h_bf    = (bf16_t*)(ws + OFF_HBF);
    bf16_t* W1t     = (bf16_t*)(ws + OFF_W1T);
    bf16_t* W2t     = (bf16_t*)(ws + OFF_W2T);
    float*  w_dense = (float*) (ws + OFF_WD);
    int*    counts  = (int*)   (ws + OFF_CNT);
    int*    lists   = (int*)   (ws + OFF_LST);

    (void)hipMemsetAsync(out, 0, (size_t)B * C * sizeof(float), stream);
    (void)hipMemsetAsync(counts, 0, SZ_CNT, stream);
    (void)hipMemsetAsync(w_dense, 0, SZ_WD, stream);

    cvt_h_kernel<<<(B * F + 255) / 256, 256, 0, stream>>>(h, h_bf, B * F);
    transpose_cvt_kernel<<<dim3(H / 32, F / 32, E), dim3(32, 8), 0, stream>>>(W1, W1t, F, H);
    transpose_cvt_kernel<<<dim3(C / 32, H / 32, E), dim3(32, 8), 0, stream>>>(W2, W2t, H, C);

    router_kernel<<<B / 4, 128, 0, stream>>>(h, W_mu, b_mu, W_lv, b_lv,
                                             w_dense, counts, lists);

    moe_gemm_kernel<<<dim3(B / TM, E), 256, SMEM_TOTAL, stream>>>(
        h_bf, W1t, b1, W2t, b2, w_dense, counts, lists, out);
}
